// CacheAwareAttention_13692355740242
// MI455X (gfx1250) — matrix operand order repaired and checked by reading, not compiled, not run on hardware
//
#include <hip/hip_runtime.h>

// Problem constants (from the reference)
#define B_      8
#define H_      32
#define S_      1024
#define D_      128
#define MAXLEN_ 2048
#define CPOS_   128
#define L_      (CPOS_ + S_)   // 1152 effective KV length
#define CK      64             // keys per chunk (128 % 64 == 0 -> never straddles cache/new)
#define NCHUNK  (L_ / CK)      // 18

typedef __attribute__((ext_vector_type(16))) _Float16 v16h;
typedef __attribute__((ext_vector_type(8)))  _Float16 v8h;
typedef __attribute__((ext_vector_type(4)))  _Float16 v4h;
typedef __attribute__((ext_vector_type(8)))  float    v8f;
typedef __attribute__((ext_vector_type(4)))  float    v4f;

__global__ __launch_bounds__(256)
void fa_cdna5_kernel(const float* __restrict__ Q,
                     const float* __restrict__ Ks,
                     const float* __restrict__ Vs,
                     const float* __restrict__ Kc,
                     const float* __restrict__ Vc,
                     float* __restrict__ O)
{
    // K chunk row-major [64 keys][128 d], V chunk transposed [128 d][64 keys],
    // per-wave P transpose scratch [16 rows][64 cols]
    __shared__ __align__(32) _Float16 kLds[CK * 128];
    __shared__ __align__(32) _Float16 vLdsT[128 * CK];
    __shared__ __align__(32) _Float16 pLds[8 * 16 * CK];

    const int tid     = threadIdx.x;
    const int wave    = tid >> 5;
    const int lane    = tid & 31;
    const int laneN   = lane & 15;    // column / N index within 16
    const int halfSel = lane >> 4;    // which 16-lane half
    const int rowOff  = halfSel * 8;  // C-layout row offset / 16-bit A-layout k-base
    const int kdb     = halfSel * 16; // unused; B reads use the rowOff k-base, same as A

    const int bh    = blockIdx.y;                 // fused (b,h) in [0,256)
    const int qbase = blockIdx.x * 128 + wave * 16;

    const float scale = 0.08838834764831845f;     // 1/sqrt(128)

    // ---- Load Q tile (16x128) into 4 A-fragments, pre-scaled ----
    // A layout (16-bit, 16x32): lane m = laneN; halves 0..7 -> k = kbase+j,
    // halves 8..15 -> k = 16 + kbase + (j-8), kbase = halfSel*8.
    const size_t qRow = ((size_t)bh * S_ + (size_t)(qbase + laneN)) * D_;
    v16h qf[4];
#pragma unroll
    for (int dc = 0; dc < 4; ++dc) {
        const float* p0 = Q + qRow + dc * 32 + rowOff;
        const float* p1 = Q + qRow + dc * 32 + 16 + rowOff;
        v4f a0 = *(const v4f*)(p0);
        v4f a1 = *(const v4f*)(p0 + 4);
        v4f b0 = *(const v4f*)(p1);
        v4f b1 = *(const v4f*)(p1 + 4);
#pragma unroll
        for (int j = 0; j < 4; ++j) {
            qf[dc][j]      = (_Float16)(a0[j] * scale);
            qf[dc][4 + j]  = (_Float16)(a1[j] * scale);
            qf[dc][8 + j]  = (_Float16)(b0[j] * scale);
            qf[dc][12 + j] = (_Float16)(b1[j] * scale);
        }
    }

    const v8f vzero = {};
    v8f acc[8];
#pragma unroll
    for (int t = 0; t < 8; ++t) acc[t] = vzero;
    float m_i[8], lsum[8];              // lsum is LANE-LOCAL partial row sum
#pragma unroll
    for (int r = 0; r < 8; ++r) { m_i[r] = -3.0e38f; lsum[r] = 0.0f; }

    _Float16* pw = pLds + wave * (16 * CK);

#pragma unroll 1
    for (int kc = 0; kc < NCHUNK; ++kc) {
        const int kv0 = kc * CK;
        __syncthreads();  // previous chunk's LDS reads complete before overwrite

        // ---- Stage K (row-major f16) and V (d-major f16) for this chunk ----
        // 64x128 f32 per tensor = 2048 float4 loads; 8 per thread.
#pragma unroll
        for (int i = 0; i < 8; ++i) {
            const int f   = tid + i * 256;   // float4 index in [0,2048)
            const int row = f >> 5;          // key row 0..63
            const int d   = (f & 31) * 4;    // d offset 0..124
            const int g   = kv0 + row;       // global kv index
            const float* krow = (g < CPOS_)
                ? (Kc + ((size_t)bh * MAXLEN_ + g) * D_)
                : (Ks + ((size_t)bh * S_ + (g - CPOS_)) * D_);
            const float* vrow = (g < CPOS_)
                ? (Vc + ((size_t)bh * MAXLEN_ + g) * D_)
                : (Vs + ((size_t)bh * S_ + (g - CPOS_)) * D_);
            v4f kd = *(const v4f*)(krow + d);
            v4f vd = *(const v4f*)(vrow + d);
            v4h kh;
#pragma unroll
            for (int j = 0; j < 4; ++j) kh[j] = (_Float16)kd[j];
            *(v4h*)&kLds[row * 128 + d] = kh;
#pragma unroll
            for (int j = 0; j < 4; ++j)
                vLdsT[(d + j) * CK + row] = (_Float16)vd[j];

            // Prefetch next chunk's K/V rows (global_prefetch_b8)
            if (kc + 1 < NCHUNK) {
                const int g2 = g + CK;
                const float* kn = (g2 < CPOS_)
                    ? (Kc + ((size_t)bh * MAXLEN_ + g2) * D_)
                    : (Ks + ((size_t)bh * S_ + (g2 - CPOS_)) * D_);
                const float* vn = (g2 < CPOS_)
                    ? (Vc + ((size_t)bh * MAXLEN_ + g2) * D_)
                    : (Vs + ((size_t)bh * S_ + (g2 - CPOS_)) * D_);
                __builtin_prefetch(kn + d, 0, 0);
                __builtin_prefetch(vn + d, 0, 0);
            }
        }
        __syncthreads();

        // ---- S = Q * K^T : four 16x16 score tiles (keys j*16..j*16+15) ----
        // B layout (16-bit, 32x16): lane col n = laneN, halves 0..7 -> k = rowOff + i,
        // halves 8..15 -> k = 16 + rowOff + (i-8) (same order as A).
        v8f s[4];
#pragma unroll
        for (int j = 0; j < 4; ++j) s[j] = vzero;
#pragma unroll
        for (int dc = 0; dc < 4; ++dc) {
#pragma unroll
            for (int j = 0; j < 4; ++j) {
                v8h bk0 = *(const v8h*)&kLds[(j * 16 + laneN) * 128 + dc * 32 + rowOff];
                v8h bk1 = *(const v8h*)&kLds[(j * 16 + laneN) * 128 + dc * 32 + 16 + rowOff];
                v16h bk = __builtin_shufflevector(bk0, bk1, 0,1,2,3,4,5,6,7,8,9,10,11,12,13,14,15);
                s[j] = __builtin_amdgcn_wmma_f32_16x16x32_f16(false, qf[dc], false, bk,
                                                              (short)0, s[j], false, false);
            }
        }

        // ---- Online softmax. Only the row-MAX needs a cross-lane butterfly
        // (rows live per-VGPR, cols per-lane; xor 1,2,4,8 stays in each 16-lane
        // half => exact row-wise reduction). Row sums stay lane-local: alpha is
        // row-uniform, so partials rescale coherently; reduce once at the end. ----
        float alpha[8];
#pragma unroll
        for (int r = 0; r < 8; ++r) {
            float mx = fmaxf(fmaxf(s[0][r], s[1][r]), fmaxf(s[2][r], s[3][r]));
            mx = fmaxf(mx, __shfl_xor(mx, 1));
            mx = fmaxf(mx, __shfl_xor(mx, 2));
            mx = fmaxf(mx, __shfl_xor(mx, 4));
            mx = fmaxf(mx, __shfl_xor(mx, 8));
            const float nm = fmaxf(m_i[r], mx);
            const float a  = __expf(m_i[r] - nm);
            float ps = 0.0f;
#pragma unroll
            for (int j = 0; j < 4; ++j) {
                const float p = __expf(s[j][r] - nm);
                s[j][r] = p;
                ps += p;
            }
            lsum[r]  = lsum[r] * a + ps;   // lane-local partial
            m_i[r]   = nm;
            alpha[r] = a;
        }
#pragma unroll
        for (int t = 0; t < 8; ++t)
#pragma unroll
            for (int r = 0; r < 8; ++r)
                acc[t][r] *= alpha[r];

        // ---- Transpose P (C layout -> A layout) via wave-private LDS.
        // Same-wave DS ops are in-order; no barrier needed. ----
#pragma unroll
        for (int r = 0; r < 8; ++r) {
            const int row = r + rowOff;
#pragma unroll
            for (int j = 0; j < 4; ++j)
                pw[row * CK + j * 16 + laneN] = (_Float16)s[j][r];
        }
        v16h ap[2];
#pragma unroll
        for (int h = 0; h < 2; ++h) {
            v8h plo = *(const v8h*)&pw[laneN * CK + h * 32 + rowOff];
            v8h phi = *(const v8h*)&pw[laneN * CK + h * 32 + 16 + rowOff];
#pragma unroll
            for (int j = 0; j < 8; ++j) { ap[h][j] = plo[j]; ap[h][8 + j] = phi[j]; }
        }

        // ---- O += P * V : 8 d-tiles x 2 k-steps. Vt gives contiguous B reads. ----
#pragma unroll
        for (int t = 0; t < 8; ++t) {
#pragma unroll
            for (int h = 0; h < 2; ++h) {
                v8h bv0 = *(const v8h*)&vLdsT[(t * 16 + laneN) * CK + h * 32 + rowOff];
                v8h bv1 = *(const v8h*)&vLdsT[(t * 16 + laneN) * CK + h * 32 + 16 + rowOff];
                v16h bv = __builtin_shufflevector(bv0, bv1, 0,1,2,3,4,5,6,7,8,9,10,11,12,13,14,15);
                acc[t] = __builtin_amdgcn_wmma_f32_16x16x32_f16(false, ap[h], false, bv,
                                                                (short)0, acc[t], false, false);
            }
        }
    }

    // ---- Epilogue: single butterfly per row to finish the sum, then store ----
#pragma unroll
    for (int r = 0; r < 8; ++r) {
        float rs = lsum[r];
        rs += __shfl_xor(rs, 1);
        rs += __shfl_xor(rs, 2);
        rs += __shfl_xor(rs, 4);
        rs += __shfl_xor(rs, 8);
        const float inv = 1.0f / rs;
        const size_t orow = ((size_t)bh * S_ + (size_t)(qbase + r + rowOff)) * D_;
#pragma unroll
        for (int t = 0; t < 8; ++t)
            O[orow + t * 16 + laneN] = acc[t][r] * inv;
    }
}

extern "C" void kernel_launch(void* const* d_in, const int* in_sizes, int n_in,
                              void* d_out, int out_size, void* d_ws, size_t ws_size,
                              hipStream_t stream)
{
    (void)in_sizes; (void)n_in; (void)out_size; (void)d_ws; (void)ws_size;
    const float* q  = (const float*)d_in[0];   // query_states  (B,H,S,D)
    const float* ks = (const float*)d_in[1];   // key_states    (B,H,S,D)
    const float* vs = (const float*)d_in[2];   // value_states  (B,H,S,D)
    const float* kc = (const float*)d_in[3];   // key_cache     (B,H,MAX_LEN,D)
    const float* vc = (const float*)d_in[4];   // value_cache   (B,H,MAX_LEN,D)
    // d_in[5] = cache_position (int, fixed 128 per the reference)
    float* out = (float*)d_out;

    dim3 grid(S_ / 128, B_ * H_);   // (8, 256)
    dim3 block(256);                // 8 wave32 waves
    fa_cdna5_kernel<<<grid, block, 0, stream>>>(q, ks, vs, kc, vc, out);
}
